// GraphAttentionLayer_9302899163408
// MI455X (gfx1250) — matrix operand order repaired and checked by reading, not compiled, not run on hardware
//
#include <hip/hip_runtime.h>

typedef __bf16 bf16_t;
typedef __attribute__((ext_vector_type(16))) __bf16 v16bf;
typedef __attribute__((ext_vector_type(8)))  float  v8f;

constexpr int   GN    = 8192;
constexpr int   GFIN  = 512;
constexpr int   GFOUT = 256;
constexpr float GALPHA = 0.2f;
constexpr float GL2E   = 1.4426950408889634f;

__device__ __forceinline__ void ld4i(const int* p, int* d) {
  int4 t = *(const int4*)p; d[0] = t.x; d[1] = t.y; d[2] = t.z; d[3] = t.w;
}
__device__ __forceinline__ void ld4f(const float* p, float* d) {
  float4 t = *(const float4*)p; d[0] = t.x; d[1] = t.y; d[2] = t.z; d[3] = t.w;
}

// B-operand swizzled index for h: row j (K-dim), col c (N-dim).
// Tile = (j/32, c/16); within tile: lane = (c&15) + 16*((j>>3)&1), element = (j&7) + 8*((j>>4)&1).
__device__ __forceinline__ size_t bswz_idx(int j, int c) {
  return ((size_t)((j >> 5) * 16 + (c >> 4))) * 512 +
         (size_t)(((c & 15) + ((j & 8) << 1)) * 16 + (j & 7) + ((j & 16) >> 1));
}

// ---------------------------------------------------------------------------
// K1: h = input @ W (bf16 WMMA, f32 accum). Writes h_f32 (for s1/s2) and
// h in pre-swizzled WMMA-B layout (for K4's single-load B tiles).
// Block: 8 waves, 16 rows x 256 cols; wave owns two 16x16 col tiles.
// ---------------------------------------------------------------------------
__global__ __launch_bounds__(256) void gat_gemm_h(
    const float* __restrict__ X, const float* __restrict__ W,
    float* __restrict__ hf, bf16_t* __restrict__ hswz) {
  const int tid  = threadIdx.x;
  const int wave = tid >> 5, lane = tid & 31;
  const int r    = lane & 15, hl = lane >> 4;
  const int ibase = blockIdx.x * 16;
  const int n0    = wave * 32 + r;          // first col (tile wave*2)
  const int ahalf = hl ? 8 : 0;
  const int bhalf = hl ? 16 : 0;

  v8f acc0 = {}, acc1 = {};
  const float* row = X + (size_t)(ibase + r) * GFIN;

  for (int kb = 0; kb < GFIN; kb += 32) {
    v16bf a;
    {
      const float4* rp  = (const float4*)(row + kb + ahalf);
      const float4* rp2 = (const float4*)(row + kb + ahalf + 16);
      float4 f0 = rp[0],  f1 = rp[1];
      float4 f2 = rp2[0], f3 = rp2[1];
      a[0]=(bf16_t)f0.x; a[1]=(bf16_t)f0.y; a[2]=(bf16_t)f0.z; a[3]=(bf16_t)f0.w;
      a[4]=(bf16_t)f1.x; a[5]=(bf16_t)f1.y; a[6]=(bf16_t)f1.z; a[7]=(bf16_t)f1.w;
      a[8]=(bf16_t)f2.x; a[9]=(bf16_t)f2.y; a[10]=(bf16_t)f2.z; a[11]=(bf16_t)f2.w;
      a[12]=(bf16_t)f3.x; a[13]=(bf16_t)f3.y; a[14]=(bf16_t)f3.z; a[15]=(bf16_t)f3.w;
    }
    v16bf b0, b1;
#pragma unroll
    for (int e = 0; e < 16; ++e) {
      const float* wrow = W + (size_t)(kb + ahalf + e + (e & 8)) * GFOUT;
      b0[e] = (bf16_t)wrow[n0];
      b1[e] = (bf16_t)wrow[n0 + 16];
    }
    acc0 = __builtin_amdgcn_wmma_f32_16x16x32_bf16(false, a, false, b0,
                                                   (short)0, acc0, false, false);
    acc1 = __builtin_amdgcn_wmma_f32_16x16x32_bf16(false, a, false, b1,
                                                   (short)0, acc1, false, false);
  }
#pragma unroll
  for (int rr = 0; rr < 8; ++rr) {
    const int orow = ibase + rr + 8 * hl;
    const size_t base = (size_t)orow * GFOUT;
    hf[base + n0]      = acc0[rr];
    hf[base + n0 + 16] = acc1[rr];
    hswz[bswz_idx(orow, n0)]      = (bf16_t)acc0[rr];
    hswz[bswz_idx(orow, n0 + 16)] = (bf16_t)acc1[rr];
  }
}

// ---------------------------------------------------------------------------
// K2: s1[i] = h[i,:].a[:F], s2[i] = h[i,:].a[F:]. One wave per row.
// ---------------------------------------------------------------------------
__global__ __launch_bounds__(256) void gat_s12(
    const float* __restrict__ hf, const float* __restrict__ avec,
    float* __restrict__ s1, float* __restrict__ s2) {
  const int wave = threadIdx.x >> 5, lane = threadIdx.x & 31;
  const int row  = blockIdx.x * 8 + wave;
  const float* hr = hf + (size_t)row * GFOUT;
  float v1 = 0.f, v2 = 0.f;
  for (int c = lane; c < GFOUT; c += 32) {
    const float hv = hr[c];
    v1 += hv * avec[c];
    v2 += hv * avec[GFOUT + c];
  }
  for (int o = 16; o > 0; o >>= 1) {
    v1 += __shfl_down(v1, o, 32);
    v2 += __shfl_down(v2, o, 32);
  }
  if (lane == 0) { s1[row] = v1; s2[row] = v2; }
}

// ---------------------------------------------------------------------------
// K3: per-row masked-softmax stats (online max/sum). One block per row,
// int4/float4 vectorized coalesced adj stream.
// ---------------------------------------------------------------------------
__global__ __launch_bounds__(256) void gat_rowstats(
    const int* __restrict__ adj, const float* __restrict__ s1,
    const float* __restrict__ s2, float* __restrict__ mrow,
    float* __restrict__ zinv) {
  __shared__ float lm[256], lz[256];
  const int row = blockIdx.x, tid = threadIdx.x;
  const float s1v = s1[row];
  const int* arow = adj + (size_t)row * GN;
  float m = -3.4e38f, z = 0.f;
  for (int j0 = tid * 4; j0 < GN; j0 += 1024) {
    int   av[4]; ld4i(arow + j0, av);
    float sv[4]; ld4f(s2 + j0, sv);
#pragma unroll
    for (int q = 0; q < 4; ++q) {
      if (av[q] > 0) {
        float e = s1v + sv[q];
        e = fmaxf(e, GALPHA * e);
        if (e > m) {
          z = z * __builtin_amdgcn_exp2f((m - e) * GL2E) + 1.f;
          m = e;
        } else {
          z += __builtin_amdgcn_exp2f((e - m) * GL2E);
        }
      }
    }
  }
  lm[tid] = m; lz[tid] = z;
  __syncthreads();
  for (int s = 128; s > 0; s >>= 1) {
    if (tid < s) {
      const float m1 = lm[tid], z1 = lz[tid];
      const float m2 = lm[tid + s], z2 = lz[tid + s];
      const float M = fmaxf(m1, m2);
      lz[tid] = z1 * __builtin_amdgcn_exp2f((m1 - M) * GL2E) +
                z2 * __builtin_amdgcn_exp2f((m2 - M) * GL2E);
      lm[tid] = M;
    }
    __syncthreads();
  }
  if (tid == 0) {
    mrow[row] = lm[0] * GL2E;
    zinv[row] = 1.f / fmaxf(lz[0], 1e-30f);
  }
}

// ---------------------------------------------------------------------------
// K4: h' = softmax(mask(leakyrelu(s1_i + s2_j))) @ h, fused.
// No LDS, no barriers. Wave = 2 row-tiles x 4 col-tiles (32 rows x 64 cols).
// Block = 8 waves = 64 rows x 256 cols. Grid = N/64 = 128 blocks.
// adj: one 128B line per row per 32-chunk, fully consumed by the wave.
// B: single v16bf load per tile from pre-swizzled h (2x global_load_b128).
// ---------------------------------------------------------------------------
__global__ __launch_bounds__(256) void gat_attn(
    const int* __restrict__ adj, const bf16_t* __restrict__ hswz,
    const float* __restrict__ s1g, const float* __restrict__ s2g,
    const float* __restrict__ mrowg, const float* __restrict__ zinvg,
    float* __restrict__ out) {
  const int tid  = threadIdx.x;
  const int wave = tid >> 5, lane = tid & 31;
  const int r    = lane & 15, hl = lane >> 4;
  const int rg   = wave >> 2;            // row group (0..1)
  const int cg   = wave & 3;             // col group (0..3)
  const int rbase = blockIdx.x * 64 + rg * 32;
  const int cbase = cg * 64;
  const int ahalf = hl ? 8 : 0;

  const int  row0  = rbase + r;          // A row, tile 0
  const int  row1  = rbase + 16 + r;     // A row, tile 1
  const int* arow0 = adj + (size_t)row0 * GN;
  const int* arow1 = adj + (size_t)row1 * GN;
  const float s1a0 = s1g[row0], mv0 = mrowg[row0];
  const float s1a1 = s1g[row1], mv1 = mrowg[row1];

  v8f acc[2][4] = {};

  for (int jb = 0; jb < GN; jb += 32) {
    // ---- gather adj + s2 for this lane's K pattern: {ahalf+0..7, 16+ahalf+0..7}
    int av0[16], av1[16]; float sv[16];
    ld4i(arow0 + jb + ahalf,      av0);      ld4i(arow0 + jb + ahalf + 4,  av0 + 4);
    ld4i(arow0 + jb + ahalf + 16, av0 + 8);  ld4i(arow0 + jb + ahalf + 20, av0 + 12);
    ld4i(arow1 + jb + ahalf,      av1);      ld4i(arow1 + jb + ahalf + 4,  av1 + 4);
    ld4i(arow1 + jb + ahalf + 16, av1 + 8);  ld4i(arow1 + jb + ahalf + 20, av1 + 12);
    ld4f(s2g + jb + ahalf,      sv);         ld4f(s2g + jb + ahalf + 4,  sv + 4);
    ld4f(s2g + jb + ahalf + 16, sv + 8);     ld4f(s2g + jb + ahalf + 20, sv + 12);

    // ---- build P tiles (A operands); mask folded into exp2 argument ----
    v16bf A0, A1;
#pragma unroll
    for (int e = 0; e < 16; ++e) {
      float e0 = s1a0 + sv[e];
      e0 = fmaxf(e0, GALPHA * e0);
      float t0 = (av0[e] > 0) ? (e0 * GL2E - mv0) : -1e30f;
      A0[e] = (bf16_t)__builtin_amdgcn_exp2f(t0);
      float e1 = s1a1 + sv[e];
      e1 = fmaxf(e1, GALPHA * e1);
      float t1 = (av1[e] > 0) ? (e1 * GL2E - mv1) : -1e30f;
      A1[e] = (bf16_t)__builtin_amdgcn_exp2f(t1);
    }

    // ---- 4 B tiles (swizzled, one v16bf load each), 8 WMMAs ----
    const size_t tbase = (size_t)((jb >> 5) * 16 + cg * 4) * 512 + (size_t)lane * 16;
#pragma unroll
    for (int nt = 0; nt < 4; ++nt) {
      v16bf B = *(const v16bf*)(hswz + tbase + (size_t)nt * 512);
      acc[0][nt] = __builtin_amdgcn_wmma_f32_16x16x32_bf16(
          false, A0, false, B, (short)0, acc[0][nt], false, false);
      acc[1][nt] = __builtin_amdgcn_wmma_f32_16x16x32_bf16(
          false, A1, false, B, (short)0, acc[1][nt], false, false);
    }

    if (jb + 32 < GN) {
      __builtin_prefetch(arow0 + jb + 32 + ahalf, 0, 1);
      __builtin_prefetch(arow1 + jb + 32 + ahalf, 0, 1);
    }
  }

  // ---- epilogue: 1/Z scale, ELU, store ----
#pragma unroll
  for (int mt = 0; mt < 2; ++mt) {
#pragma unroll
    for (int rr = 0; rr < 8; ++rr) {
      const int   orow = rbase + mt * 16 + rr + 8 * hl;
      const float zi   = zinvg[orow];
      const size_t base = (size_t)orow * GFOUT + cbase + r;
#pragma unroll
      for (int nt = 0; nt < 4; ++nt) {
        float v = acc[mt][nt][rr] * zi;
        v = v > 0.f ? v : __builtin_amdgcn_exp2f(v * GL2E) - 1.f;
        out[base + nt * 16] = v;
      }
    }
  }
}

// ---------------------------------------------------------------------------
extern "C" void kernel_launch(void* const* d_in, const int* in_sizes, int n_in,
                              void* d_out, int out_size, void* d_ws, size_t ws_size,
                              hipStream_t stream) {
  const float* X   = (const float*)d_in[0];   // [8192,512]
  const int*   adj = (const int*)d_in[1];     // [8192,8192]
  const float* W   = (const float*)d_in[2];   // [512,256]
  const float* av  = (const float*)d_in[3];   // [512,1]
  float*       out = (float*)d_out;           // [8192,256]

  char* ws = (char*)d_ws;
  float*  hf   = (float*)ws;                           // 8 MB
  bf16_t* hswz = (bf16_t*)(ws + (size_t)8388608);      // 4 MB, B-swizzled
  float*  s1   = (float*)(ws + (size_t)12582912);
  float*  s2   = s1 + GN;
  float*  mrow = s2 + GN;
  float*  zinv = mrow + GN;

  gat_gemm_h  <<<GN / 16, 256, 0, stream>>>(X, W, hf, hswz);
  gat_s12     <<<GN / 8,  256, 0, stream>>>(hf, av, s1, s2);
  gat_rowstats<<<GN,      256, 0, stream>>>(adj, s1, s2, mrow, zinv);
  gat_attn    <<<GN / 64, 256, 0, stream>>>(adj, hswz, s1, s2, mrow, zinv, out);
}
